// Cross_Attention_69801808495347
// MI455X (gfx1250) — compile-verified, matrix operand order checked by reading, not run on hardware
//
#include <hip/hip_runtime.h>
#include <math.h>

#define D_MODEL  1024
#define NUM_HEAD 16
#define HEAD_DIM 64
#define LSEQ     2048           // Lt == Ls == 2048 (compile-time: no int division)
#define LSEQ_SH  11             // log2(LSEQ)
#define AT_WAVES 2              // waves per attention block

// CDNA5 async global->LDS staging (ASYNCcnt path), with safe fallback.
#if defined(__has_builtin)
#if __has_builtin(__builtin_amdgcn_global_load_async_to_lds_b128) && \
    __has_builtin(__builtin_amdgcn_s_wait_asynccnt)
#define ASYNC_OK 1
#endif
#endif
#ifndef ASYNC_OK
#define ASYNC_OK 0
#endif

#define GLOBAL_AS __attribute__((address_space(1)))
#define LDS_AS    __attribute__((address_space(3)))

typedef __attribute__((ext_vector_type(16))) __bf16 v16bf;
typedef __attribute__((ext_vector_type(8)))  __bf16 v8bf;
typedef __attribute__((ext_vector_type(8)))  float  v8f;
typedef __attribute__((ext_vector_type(4)))  float  v4f;
typedef __attribute__((ext_vector_type(4)))  int    v4i;

// K-index mapping for 16-bit A/B fragments (ISA 7.12.2, 16-bit A 16x32):
// element e of the v16bf fragment, for this lane's half (lane>=16 -> half=1):
//   e in [0..7]  -> k = 8*half + e
//   e in [8..15] -> k = 16 + 8*half + (e-8)
// i.e. two contiguous 8-element runs -> two 16B vector loads.
__device__ __forceinline__ v8f vzero8() {
  v8f z = {0.f, 0.f, 0.f, 0.f, 0.f, 0.f, 0.f, 0.f};
  return z;
}

__device__ __forceinline__ v16bf load_frag_row_bf16(const __bf16* rowp, int half) {
  const int off = half * 8;
  v8bf lo = *reinterpret_cast<const v8bf*>(rowp + off);
  v8bf hi = *reinterpret_cast<const v8bf*>(rowp + 16 + off);
  v16bf r;
#pragma unroll
  for (int i = 0; i < 8; ++i) { r[i] = lo[i]; r[i + 8] = hi[i]; }
  return r;
}

#if ASYNC_OK
// One wave-wide 512B async copy: 32 lanes x 16B, global -> LDS.
__device__ __forceinline__ void async_copy16(const __bf16* g, __bf16* l) {
  __builtin_amdgcn_global_load_async_to_lds_b128(
      (GLOBAL_AS v4i*)g, (LDS_AS v4i*)l, 0, 0);
}

// Stage one 32-key K tile (32x64, contiguous 4KB) and one V tile (64x32,
// rows strided by LSEQ) into this wave's LDS buffers: 16 async ops total.
__device__ __forceinline__ void stage_kv(const __bf16* kwave, const __bf16* vwave,
                                         __bf16* kb, __bf16* vb, int s0, int lane) {
  const __bf16* kg = kwave + (size_t)s0 * HEAD_DIM + lane * 8;
  __bf16* kl = kb + lane * 8;
#pragma unroll
  for (int j = 0; j < 8; ++j)
    async_copy16(kg + j * 256, kl + j * 256);           // 512B per op

  const int dl = lane >> 2, part = lane & 3;            // 8 dims x 4 parts per op
  const __bf16* vg = vwave + (size_t)dl * LSEQ + s0 + part * 8;
  __bf16* vl = vb + dl * 32 + part * 8;
#pragma unroll
  for (int j = 0; j < 8; ++j)
    async_copy16(vg + (size_t)(j * 8) * LSEQ, vl + j * 8 * 32);
}
#endif

// ---------------------------------------------------------------------------
// Elementwise fp32 -> bf16 conversion (one-shot; removes per-tile re-convert).
// ---------------------------------------------------------------------------
__global__ void __launch_bounds__(256)
cvt_kernel(const float* __restrict__ in, __bf16* __restrict__ out, int n8) {
  const int i = blockIdx.x * blockDim.x + threadIdx.x;
  if (i >= n8) return;
  const v4f a = reinterpret_cast<const v4f*>(in)[2 * i];
  const v4f b = reinterpret_cast<const v4f*>(in)[2 * i + 1];
  v8bf o;
#pragma unroll
  for (int j = 0; j < 4; ++j) { o[j] = (__bf16)a[j]; o[j + 4] = (__bf16)b[j]; }
  reinterpret_cast<v8bf*>(out)[i] = o;
}

// ---------------------------------------------------------------------------
// Weight transpose+convert: Wt_bf16[n][k] = (bf16)W_f32[k][n]  (one-shot).
// Coalesced fp32 reads across lanes; 16B bf16 row writes per lane.
// ---------------------------------------------------------------------------
__global__ void __launch_bounds__(32)
wtr_kernel(const float* __restrict__ W, __bf16* __restrict__ Wt) {
  const int n = blockIdx.x * 32 + threadIdx.x;
  for (int k = 0; k < D_MODEL; k += 8) {
    v8bf o;
#pragma unroll
    for (int i = 0; i < 8; ++i)
      o[i] = (__bf16)W[(size_t)(k + i) * D_MODEL + n];
    *reinterpret_cast<v8bf*>(&Wt[(size_t)n * D_MODEL + k]) = o;
  }
}

// ---------------------------------------------------------------------------
// Unified WMMA GEMM: [M x 1024](bf16) @ Wt[1024 x 1024](bf16, pre-transposed).
// One wave computes a 16x64 tile; A-fragment reused across 4 WMMAs per chunk;
// all operand loads are 16B vector loads.
// MODE 0: store bf16 per-head row-major  [B,H,LSEQ,64]   (Q and K)
// MODE 2: store bf16 per-head transposed [B,H,64,LSEQ]   (V)
// MODE 3: store fp32 row-major [M x 1024]                (final output)
// ---------------------------------------------------------------------------
template <int MODE>
__global__ void __launch_bounds__(32)
gemm_kernel(const __bf16* __restrict__ A, const __bf16* __restrict__ Wt,
            __bf16* __restrict__ outb, float* __restrict__ outf) {
  const int lane = threadIdx.x & 31;
  const int half = lane >> 4;
  const int l16  = lane & 15;
  const int row0 = blockIdx.y * 16;
  const int col0 = blockIdx.x * 64;
  const int arow = row0 + l16;

  v8f acc[4];
#pragma unroll
  for (int t = 0; t < 4; ++t) acc[t] = vzero8();

  for (int kc = 0; kc < D_MODEL; kc += 32) {
    const v16bf af = load_frag_row_bf16(A + (size_t)arow * D_MODEL + kc, half);
#pragma unroll
    for (int t = 0; t < 4; ++t) {
      const int bcol = col0 + t * 16 + l16;
      const v16bf bf_ = load_frag_row_bf16(Wt + (size_t)bcol * D_MODEL + kc, half);
      acc[t] = __builtin_amdgcn_wmma_f32_16x16x32_bf16(false, af, false, bf_,
                                                       (short)0, acc[t], false, false);
    }
  }

  // C/D layout: lane holds column l16; VGPR r holds row r (+8 for upper half)
#pragma unroll
  for (int t = 0; t < 4; ++t) {
    const int bcol = col0 + t * 16 + l16;
    const int h = bcol >> 6;          // / HEAD_DIM
    const int d = bcol & (HEAD_DIM - 1);
#pragma unroll
    for (int r = 0; r < 8; ++r) {
      const int grow = row0 + r + half * 8;       // global row in [0, B*LSEQ)
      if (MODE == 3) {
        outf[(size_t)grow * D_MODEL + bcol] = acc[t][r];
      } else {
        const int b  = grow >> LSEQ_SH;
        const int tq = grow & (LSEQ - 1);
        size_t idx;
        if (MODE == 2) idx = ((size_t)(b * NUM_HEAD + h) * HEAD_DIM + d) * LSEQ + tq;
        else           idx = ((size_t)(b * NUM_HEAD + h) * LSEQ + tq) * HEAD_DIM + d;
        outb[idx] = (__bf16)acc[t][r];
      }
    }
  }
}

// ---------------------------------------------------------------------------
// Fused flash attention, S-transposed formulation.
// Per wave: one 16-query tile of one (b,h); keys streamed 32 at a time with
// double-buffered async global->LDS staging (ASYNCcnt).
// S^T = K Q^T  (A = K rows, B = Q rows) puts the QUERY on the lane axis:
//   - softmax stats are per-lane scalars (one shfl_xor(16) per reduction)
//   - pos/mask bias loads are per-lane contiguous (v4f vector loads)
//   - P^T writes to LDS are contiguous (2 x ds_store_b128)
// plds is wave-private and LDS ops are wave-ordered -> no block barrier.
// ---------------------------------------------------------------------------
__global__ void __launch_bounds__(32 * AT_WAVES)
attn_kernel(const __bf16* __restrict__ Q, const __bf16* __restrict__ Km,
            const __bf16* __restrict__ Vt, const float* __restrict__ pos,
            const float* __restrict__ mask, __bf16* __restrict__ Ocat) {
#if ASYNC_OK
  __shared__ __attribute__((aligned(16))) __bf16 kstage[AT_WAVES][2][32 * HEAD_DIM];
  __shared__ __attribute__((aligned(16))) __bf16 vstage[AT_WAVES][2][HEAD_DIM * 32];
#endif
  __shared__ __attribute__((aligned(16))) __bf16 plds[AT_WAVES][16 * 32];

  const int wv   = threadIdx.x >> 5;
  const int lane = threadIdx.x & 31;
  const int half = lane >> 4;
  const int l16  = lane & 15;
  const int bh   = blockIdx.y;              // b*NUM_HEAD + h
  const int b    = bh >> 4;
  const int h    = bh & 15;
  const int qrow0 = (blockIdx.x * AT_WAVES + wv) * 16;
  const float scale = 0.125f;               // 1/sqrt(HEAD_DIM)

  // Q fragments (16x64 -> two 16x32), resident for the whole loop.
  const __bf16* qbase = Q + ((size_t)bh * LSEQ + (qrow0 + l16)) * HEAD_DIM;
  const v16bf qa0 = load_frag_row_bf16(qbase, half);
  const v16bf qa1 = load_frag_row_bf16(qbase + 32, half);

  // This lane's query row for the bias streams (per-lane scalars).
  const size_t posrow  = ((size_t)h * LSEQ + qrow0 + l16) * LSEQ;  // pos[1,H,Lt,Ls]
  const size_t maskrow = (size_t)(qrow0 + l16) * LSEQ;             // mask[Lt,Ls]

  const __bf16* kwave = Km + (size_t)bh * LSEQ * HEAD_DIM;
  const __bf16* vwave = Vt + (size_t)bh * HEAD_DIM * LSEQ;

  float m_i = -3.0e38f, l_i = 0.f;          // per-lane (query) running stats
  v8f acc[4];
#pragma unroll
  for (int t = 0; t < 4; ++t) acc[t] = vzero8();

#if ASYNC_OK
  stage_kv(kwave, vwave, kstage[wv][0], vstage[wv][0], 0, lane);
#endif

  for (int s0 = 0; s0 < LSEQ; s0 += 32) {
    const int cur = (s0 >> 5) & 1;
    (void)cur;
#if ASYNC_OK
    if (s0 + 32 < LSEQ) {
      stage_kv(kwave, vwave, kstage[wv][cur ^ 1], vstage[wv][cur ^ 1], s0 + 32, lane);
      __builtin_amdgcn_s_wait_asynccnt(16);   // in-order: previous tile landed
    } else {
      __builtin_amdgcn_s_wait_asynccnt(0);
    }
#endif
    // Prefetch next key-block segment of this lane's pos row (268MB stream).
    if (s0 + 32 < LSEQ)
      __builtin_prefetch(pos + posrow + s0 + 32, 0, 1);

    // ---- S^T = K Q^T for two 16-key sub-tiles, fused bias/mask ----
    // C layout of S^T: lane = query l16; VGPR r = key (sub*16 + 8*half + r).
    float stile[2][8];
#pragma unroll
    for (int sub = 0; sub < 2; ++sub) {
#if ASYNC_OK
      const __bf16* kbase = &kstage[wv][cur][(sub * 16 + l16) * HEAD_DIM];
#else
      const __bf16* kbase = kwave + (size_t)(s0 + sub * 16 + l16) * HEAD_DIM;
#endif
      v16bf kf0 = load_frag_row_bf16(kbase, half);       // A-fragment: key rows
      v16bf kf1 = load_frag_row_bf16(kbase + 32, half);
      v8f sT = __builtin_amdgcn_wmma_f32_16x16x32_bf16(false, kf1, false, qa1,
                                                       (short)0, vzero8(), false, false);
      sT = __builtin_amdgcn_wmma_f32_16x16x32_bf16(false, kf0, false, qa0,
                                                   (short)0, sT, false, false);
      const int kb0 = s0 + sub * 16 + 8 * half;          // first key this lane holds
      const v4f p0 = *reinterpret_cast<const v4f*>(pos  + posrow  + kb0);
      const v4f p1 = *reinterpret_cast<const v4f*>(pos  + posrow  + kb0 + 4);
      const v4f g0 = *reinterpret_cast<const v4f*>(mask + maskrow + kb0);
      const v4f g1 = *reinterpret_cast<const v4f*>(mask + maskrow + kb0 + 4);
#pragma unroll
      for (int r = 0; r < 4; ++r) {
        stile[sub][r]     = sT[r]     * scale + p0[r] + g0[r];
        stile[sub][r + 4] = sT[r + 4] * scale + p1[r] + g1[r];
      }
    }

    // ---- per-lane online softmax (16 keys in-lane + partner half-lane) ----
    float pm = stile[0][0];
#pragma unroll
    for (int sub = 0; sub < 2; ++sub)
#pragma unroll
      for (int r = 0; r < 8; ++r) pm = fmaxf(pm, stile[sub][r]);
    pm = fmaxf(pm, __shfl_xor(pm, 16, 32));
    const float mnew  = fmaxf(m_i, pm);
    const float alpha = __expf(m_i - mnew);
    float rs = 0.f;
#pragma unroll
    for (int sub = 0; sub < 2; ++sub)
#pragma unroll
      for (int r = 0; r < 8; ++r) {
        stile[sub][r] = __expf(stile[sub][r] - mnew);
        rs += stile[sub][r];
      }
    rs += __shfl_xor(rs, 16, 32);
    l_i = l_i * alpha + rs;
    m_i = mnew;

    // ---- broadcast row alphas (query r+8*half lives in lane r+8*half) ----
    float arow[8];
#pragma unroll
    for (int r = 0; r < 8; ++r) arow[r] = __shfl(alpha, r + 8 * half, 32);
#pragma unroll
    for (int t = 0; t < 4; ++t)
#pragma unroll
      for (int r = 0; r < 8; ++r) acc[t][r] *= arow[r];

    // ---- P^T -> plds[query][key]: contiguous 16B per sub (wave-private) ----
#pragma unroll
    for (int sub = 0; sub < 2; ++sub) {
      v8bf pw;
#pragma unroll
      for (int i = 0; i < 8; ++i) pw[i] = (__bf16)stile[sub][i];
      *reinterpret_cast<v8bf*>(&plds[wv][l16 * 32 + sub * 16 + 8 * half]) = pw;
    }
    const v16bf pa = load_frag_row_bf16(&plds[wv][l16 * 32], half);

    // ---- O += P V ----
#pragma unroll
    for (int t = 0; t < 4; ++t) {
#if ASYNC_OK
      const __bf16* vbase = &vstage[wv][cur][(t * 16 + l16) * 32];
#else
      const __bf16* vbase = vwave + (size_t)(t * 16 + l16) * LSEQ + s0;
#endif
      v16bf vb = load_frag_row_bf16(vbase, half);
      acc[t] = __builtin_amdgcn_wmma_f32_16x16x32_bf16(false, pa, false, vb,
                                                       (short)0, acc[t], false, false);
    }
  }

  // ---- normalize and store score_cat[b][t][h*64+d] as bf16 ----
  const float linv = 1.0f / l_i;
  float lrow[8];
#pragma unroll
  for (int r = 0; r < 8; ++r) lrow[r] = __shfl(linv, r + 8 * half, 32);
#pragma unroll
  for (int r = 0; r < 8; ++r) {
    const int row = qrow0 + r + half * 8;
#pragma unroll
    for (int t = 0; t < 4; ++t) {
      Ocat[((size_t)(b * LSEQ + row)) * D_MODEL + h * HEAD_DIM + t * 16 + l16] =
          (__bf16)(acc[t][r] * lrow[r]);
    }
  }
}

// ---------------------------------------------------------------------------
extern "C" void kernel_launch(void* const* d_in, const int* in_sizes, int n_in,
                              void* d_out, int out_size, void* d_ws, size_t ws_size,
                              hipStream_t stream) {
  const float* x    = (const float*)d_in[0];
  const float* memv = (const float*)d_in[1];
  const float* pos  = (const float*)d_in[2];
  const float* mask = (const float*)d_in[3];
  const float* Wq   = (const float*)d_in[4];
  const float* Wk   = (const float*)d_in[5];
  const float* Wv   = (const float*)d_in[6];
  const float* Wo   = (const float*)d_in[7];

  const int B  = in_sizes[0] / (LSEQ * D_MODEL);
  const int Mq = B * LSEQ;
  const int Ms = B * LSEQ;
  const size_t DD = (size_t)D_MODEL * D_MODEL;

  __bf16* xb  = (__bf16*)d_ws;
  __bf16* mb  = xb  + (size_t)Mq * D_MODEL;
  __bf16* wqt = mb  + (size_t)Ms * D_MODEL;
  __bf16* wkt = wqt + DD;
  __bf16* wvt = wkt + DD;
  __bf16* wot = wvt + DD;
  __bf16* qb  = wot + DD;
  __bf16* kb  = qb  + (size_t)Mq * D_MODEL;
  __bf16* vb  = kb  + (size_t)Ms * D_MODEL;
  __bf16* sc  = vb  + (size_t)Ms * D_MODEL;

  // One-shot conversions / weight transposes.
  const int n8q = Mq * D_MODEL / 8;
  const int n8s = Ms * D_MODEL / 8;
  cvt_kernel<<<(n8q + 255) / 256, 256, 0, stream>>>(x,    xb, n8q);
  cvt_kernel<<<(n8s + 255) / 256, 256, 0, stream>>>(memv, mb, n8s);
  wtr_kernel<<<D_MODEL / 32, 32, 0, stream>>>(Wq, wqt);
  wtr_kernel<<<D_MODEL / 32, 32, 0, stream>>>(Wk, wkt);
  wtr_kernel<<<D_MODEL / 32, 32, 0, stream>>>(Wv, wvt);
  wtr_kernel<<<D_MODEL / 32, 32, 0, stream>>>(Wo, wot);

  // Projections.
  dim3 gQ(D_MODEL / 64, Mq / 16);
  dim3 gS(D_MODEL / 64, Ms / 16);
  gemm_kernel<0><<<gQ, 32, 0, stream>>>(xb, wqt, qb, nullptr);
  gemm_kernel<0><<<gS, 32, 0, stream>>>(mb, wkt, kb, nullptr);
  gemm_kernel<2><<<gS, 32, 0, stream>>>(mb, wvt, vb, nullptr);

  // Fused attention.
  dim3 gA(LSEQ / 16 / AT_WAVES, B * NUM_HEAD);
  attn_kernel<<<gA, 32 * AT_WAVES, 0, stream>>>(qb, kb, vb, pos, mask, sc);

  // Output projection (fp32 out).
  dim3 gO(D_MODEL / 64, Mq / 16);
  gemm_kernel<3><<<gO, 32, 0, stream>>>(sc, wot, nullptr, (float*)d_out);
}
